// RelationFeature_70325794505464
// MI455X (gfx1250) — matrix operand order repaired and checked by reading, not compiled, not run on hardware
//
#include <hip/hip_runtime.h>

// ---------------- problem constants ----------------
#define BATCH 2
#define NN    1024          // boxes per batch
#define CC    1024          // channels
#define UU    64            // units
#define NBROW (BATCH*NN)    // 2048 rows for projections
#define NSPLIT 4            // m-range splits in combine kernel

// ---------------- types ----------------
typedef __attribute__((ext_vector_type(16))) __bf16 v16bf;
typedef __attribute__((ext_vector_type(8)))  float  v8f;
typedef __attribute__((ext_vector_type(8)))  unsigned short u16x8;
typedef __attribute__((ext_vector_type(16))) unsigned short u16x16;

union BFV { u16x16 u; v16bf b; };

__device__ __forceinline__ unsigned short f2bf(float f) {
  unsigned int u = __float_as_uint(f);
  return (unsigned short)((u + 0x7FFFu + ((u >> 16) & 1u)) >> 16); // RNE
}

// A-tile (16x32 bf16): lane l holds row (l&15); elems 0..7 = K (l>>4)*16+e,
// elems 8..15 = K (l>>4)*16+8+e  -> two contiguous 16B loads.
__device__ __forceinline__ v16bf loadA16(const unsigned short* p0, const unsigned short* p1) {
  u16x8 lo = *(const u16x8*)p0;
  u16x8 hi = *(const u16x8*)p1;
  BFV a;
  a.u = __builtin_shufflevector(lo, hi, 0,1,2,3,4,5,6,7,8,9,10,11,12,13,14,15);
  return a.b;
}
// B-tile (32x16 bf16): lane l holds col (l&15), K = (l>>4)*16 + {0..15} -> one 32B load.
__device__ __forceinline__ v16bf loadB16(const unsigned short* p) {
  BFV t; t.u = *(const u16x16*)p; return t.b;
}
__device__ __forceinline__ v8f wmma_bf16(v16bf a, v16bf b, v8f c) {
  return __builtin_amdgcn_wmma_f32_16x16x32_bf16(false, a, false, b, (short)0, c, false, false);
}

// ---------------- workspace layout (bytes) ----------------
static constexpr size_t OFF_FMBF  = 0;                                   // [2048][1024] bf16
static constexpr size_t OFF_WPACK = OFF_FMBF  + (size_t)NBROW*CC*2;      // 3 mats * [4 ut][32 kt][32 lane][16] bf16
static constexpr size_t OFF_KBF   = OFF_WPACK + (size_t)3*4*32*512*2;    // [2048][64] bf16
static constexpr size_t OFF_QBF   = OFF_KBF   + (size_t)NBROW*UU*2;      // [2048][64] bf16
static constexpr size_t OFF_VT    = OFF_QBF   + (size_t)NBROW*UU*2;      // [B][64][1024] bf16 (transposed v)
static constexpr size_t OFF_WMAT  = OFF_VT    + (size_t)BATCH*UU*NN*2;   // [B][1024][1024] bf16
static constexpr size_t OFF_GEO4  = OFF_WMAT  + (size_t)BATCH*NN*NN*2;   // [B][1024][4] f32 {cx,cy,logw,logh}
static constexpr size_t OFF_GCOEF = OFF_GEO4  + (size_t)BATCH*NN*4*4;    // 5 f32 (collapsed geometry MLP)
static constexpr size_t OFF_RSP   = OFF_GCOEF + 256;                     // [B][1024][NSPLIT] f32 rowsum partials

// ============ prep: fm f32 -> bf16 ============
__global__ void k_cvt_fm(const float* __restrict__ fm, unsigned short* __restrict__ fmb) {
  int i = (blockIdx.x * blockDim.x + threadIdx.x) * 4;
  #pragma unroll
  for (int j = 0; j < 4; ++j) fmb[i + j] = f2bf(fm[i + j]);
}

// ============ prep: pack Wk/Wq/Wv [1024][64] into exact B-tile layout ============
__global__ void k_pack_w(const float* __restrict__ Wk, const float* __restrict__ Wq,
                         const float* __restrict__ Wv, unsigned short* __restrict__ wp) {
  int kt = blockIdx.x, ut = blockIdx.y, mat = blockIdx.z;
  const float* W = (mat == 0) ? Wk : (mat == 1) ? Wq : Wv;
  int t = threadIdx.x;             // 0..511
  int lane = t >> 4, e = t & 15;
  int col = ut * 16 + (lane & 15);
  int k   = kt * 32 + ((lane >= 16) ? 16 : 0) + e;
  wp[(((size_t)(mat * 4 + ut) * 32 + kt) * 512) + t] = f2bf(W[k * UU + col]);
}

// ============ prep: per-box geometry {cx, cy, log w, log h} ============
__global__ void k_geo(const float* __restrict__ boxes, float* __restrict__ geo4) {
  int i = blockIdx.x * blockDim.x + threadIdx.x;   // 0..2047
  const float* bx = boxes + (size_t)i * 4;
  float x0 = bx[0], y0 = bx[1], x1 = bx[2], y1 = bx[3];
  float4* g = (float4*)geo4;
  g[i] = make_float4(0.5f * (x0 + x1), 0.5f * (y0 + y1), __logf(x1 - x0), __logf(y1 - y0));
}

// ============ prep: collapse geometry MLP (no activation between layers) ============
// c[j] = sum_u Wg[j][u]*Wg2[u]  (j=0..3),  c[4] = sum_u bg[u]*Wg2[u] + bg2
__global__ void k_gcoef(const float* __restrict__ Wg, const float* __restrict__ bg,
                        const float* __restrict__ Wg2, const float* __restrict__ bg2,
                        float* __restrict__ gc) {
  int t = threadIdx.x;
  if (t < 4) {
    float s = 0.f;
    for (int u = 0; u < UU; ++u) s += Wg[t * UU + u] * Wg2[u];
    gc[t] = s;
  } else if (t == 4) {
    float s = 0.f;
    for (int u = 0; u < UU; ++u) s += bg[u] * Wg2[u];
    gc[4] = s + bg2[0];
  }
}

// ============ projections: [2048,1024] @ [1024,64] for K/Q/V (WMMA bf16) ============
__global__ void k_proj(const unsigned short* __restrict__ fmb,
                       const unsigned short* __restrict__ wpack,
                       const float* __restrict__ bk, const float* __restrict__ bq,
                       const float* __restrict__ bv,
                       unsigned short* __restrict__ kbf, unsigned short* __restrict__ qbf,
                       unsigned short* __restrict__ vt) {
  int lane = threadIdx.x, strip = blockIdx.x, ut = blockIdx.y, mat = blockIdx.z;
  int g = lane >> 4, lm = lane & 15;
  const unsigned short* wp = wpack + (size_t)(mat * 4 + ut) * 32 * 512;
  int row = strip * 16 + lm;
  v8f acc = {0.f,0.f,0.f,0.f,0.f,0.f,0.f,0.f};
  #pragma unroll 8
  for (int kt = 0; kt < CC / 32; ++kt) {
    const unsigned short* pr = fmb + (size_t)row * CC + kt * 32 + g * 16;
    v16bf a = loadA16(pr, pr + 8);
    v16bf b = loadB16(wp + kt * 512 + lane * 16);
    acc = wmma_bf16(a, b, acc);
  }
  const float* bias = (mat == 0) ? bk : (mat == 1) ? bq : bv;
  float bv_ = bias[ut * 16 + lm];
  int u = ut * 16 + lm;
  if (mat < 2) {
    unsigned short* out = (mat == 0) ? kbf : qbf;
    #pragma unroll
    for (int i = 0; i < 8; ++i) {
      int r = strip * 16 + i + g * 8;
      out[(size_t)r * UU + u] = f2bf(acc[i] + bv_);
    }
  } else {
    #pragma unroll
    for (int i = 0; i < 8; ++i) {
      int r = strip * 16 + i + g * 8;
      int b_ = r >> 10, n = r & (NN - 1);
      vt[((size_t)b_ * UU + u) * NN + n] = f2bf(acc[i] + bv_);   // v transposed [B][U][N]
    }
  }
}

// ============ fused: app = k q^T / 8 (WMMA) + geometry + combine -> w (bf16), rowsum partials ============
__global__ void k_combine(const unsigned short* __restrict__ kbf,
                          const unsigned short* __restrict__ qbf,
                          const float* __restrict__ geo4, const float* __restrict__ gc,
                          unsigned short* __restrict__ wmat, float* __restrict__ rsp) {
  int lane = threadIdx.x, strip = blockIdx.x, split = blockIdx.y, b = blockIdx.z;
  int g = lane >> 4, lm = lane & 15;
  const float4* g4 = (const float4*)(geo4 + (size_t)b * NN * 4);
  float c0 = gc[0], c1 = gc[1], c2 = gc[2], c3 = gc[3], c4 = gc[4];

  // row invariants for the 8 rows this lane owns (C-layout rows i + 8*g)
  float cxr[8], cyr[8], lwr[8], lhr[8];
  #pragma unroll
  for (int i = 0; i < 8; ++i) {
    float4 r = g4[strip * 16 + i + g * 8];
    cxr[i] = r.x; cyr[i] = r.y; lwr[i] = r.z; lhr[i] = r.w;
  }
  // k rows for this strip as two A tiles (u = 0..31, 32..63), loop invariant
  const unsigned short* krow = kbf + ((size_t)(b * NN + strip * 16 + lm)) * UU;
  v16bf a0 = loadA16(krow +      g * 16, krow +  8 + g * 16);
  v16bf a1 = loadA16(krow + 32 + g * 16, krow + 40 + g * 16);

  float rs[8] = {0.f,0.f,0.f,0.f,0.f,0.f,0.f,0.f};
  for (int mt = 0; mt < (NN / NSPLIT) / 16; ++mt) {
    int mb = split * (NN / NSPLIT) + mt * 16;
    int m  = mb + lm;
    const unsigned short* qrow = qbf + ((size_t)(b * NN + m)) * UU;
    v16bf b0 = loadB16(qrow +      g * 16);
    v16bf b1 = loadB16(qrow + 32 + g * 16);
    v8f acc = {0.f,0.f,0.f,0.f,0.f,0.f,0.f,0.f};
    acc = wmma_bf16(a0, b0, acc);
    acc = wmma_bf16(a1, b1, acc);          // app*8 for 16x16 tile (C-layout)

    float4 cm = g4[m];                      // column invariants (this lane's m)
    #pragma unroll
    for (int i = 0; i < 8; ++i) {
      float dx = __logf(fmaxf(fabsf(cxr[i] - cm.x), 1e-3f)) - lwr[i];
      float dy = __logf(fmaxf(fabsf(cyr[i] - cm.y), 1e-3f)) - lhr[i];
      float gv = c0 * dx + c1 * dy + c2 * (cm.z - lwr[i]) + c3 * (cm.w - lhr[i]) + c4;
      gv = fmaxf(gv, 0.f);
      float wv = fmaxf(gv * __expf(acc[i] * 0.125f), 1e-4f);
      rs[i] += wv;
      int n = strip * 16 + i + g * 8;
      wmat[(size_t)b * NN * NN + (size_t)n * NN + m] = f2bf(wv);
    }
  }
  // deterministic cross-lane row reduction (no fp atomics)
  __shared__ float lds[8][32];
  #pragma unroll
  for (int i = 0; i < 8; ++i) lds[i][lane] = rs[i];
  __syncthreads();
  if (lane < 16) {
    int i = lane & 7, off = (lane >> 3) * 16;   // rows 0..7 from lanes 0-15; 8..15 from 16-31
    float s = 0.f;
    #pragma unroll
    for (int l = 0; l < 16; ++l) s += lds[i][off + l];
    rsp[((size_t)(b * NN + strip * 16 + lane)) * NSPLIT + split] = s;
  }
}

// ============ output: out = diag(1/rowsum) * (w @ v)  (WMMA bf16) ============
__global__ void k_out(const unsigned short* __restrict__ wmat,
                      const unsigned short* __restrict__ vt,
                      const float* __restrict__ rsp, float* __restrict__ out) {
  int lane = threadIdx.x, strip = blockIdx.x, b = blockIdx.z;
  int g = lane >> 4, lm = lane & 15;
  const unsigned short* wrow = wmat + (size_t)b * NN * NN + (size_t)(strip * 16 + lm) * NN;
  const unsigned short* vtb  = vt + (size_t)b * UU * NN;
  v8f acc[4];
  #pragma unroll
  for (int ut = 0; ut < 4; ++ut) acc[ut] = (v8f){0.f,0.f,0.f,0.f,0.f,0.f,0.f,0.f};
  for (int kt = 0; kt < NN / 32; ++kt) {
    const unsigned short* pa = wrow + kt * 32 + g * 16;
    v16bf a = loadA16(pa, pa + 8);
    #pragma unroll
    for (int ut = 0; ut < 4; ++ut) {
      v16bf bt = loadB16(vtb + (size_t)(ut * 16 + lm) * NN + kt * 32 + g * 16);
      acc[ut] = wmma_bf16(a, bt, acc[ut]);
    }
  }
  float inv[8];
  const float4* rp4 = (const float4*)rsp;
  #pragma unroll
  for (int i = 0; i < 8; ++i) {
    int n = strip * 16 + i + g * 8;
    float4 p = rp4[b * NN + n];
    inv[i] = 1.f / (p.x + p.y + p.z + p.w);
  }
  float* ob = out + (size_t)b * NN * UU;
  #pragma unroll
  for (int ut = 0; ut < 4; ++ut) {
    int u = ut * 16 + lm;
    #pragma unroll
    for (int i = 0; i < 8; ++i) {
      int n = strip * 16 + i + g * 8;
      ob[(size_t)n * UU + u] = acc[ut][i] * inv[i];
    }
  }
}

extern "C" void kernel_launch(void* const* d_in, const int* in_sizes, int n_in,
                              void* d_out, int out_size, void* d_ws, size_t ws_size,
                              hipStream_t stream) {
  (void)in_sizes; (void)n_in; (void)out_size; (void)ws_size;
  const float* fm    = (const float*)d_in[0];
  const float* boxes = (const float*)d_in[1];
  const float* Wg    = (const float*)d_in[2];
  const float* bg    = (const float*)d_in[3];
  const float* Wg2   = (const float*)d_in[4];
  const float* bg2   = (const float*)d_in[5];
  const float* Wk    = (const float*)d_in[6];
  const float* bk    = (const float*)d_in[7];
  const float* Wq    = (const float*)d_in[8];
  const float* bq    = (const float*)d_in[9];
  const float* Wv    = (const float*)d_in[10];
  const float* bv    = (const float*)d_in[11];

  char* ws = (char*)d_ws;
  unsigned short* fmb   = (unsigned short*)(ws + OFF_FMBF);
  unsigned short* wpack = (unsigned short*)(ws + OFF_WPACK);
  unsigned short* kbf   = (unsigned short*)(ws + OFF_KBF);
  unsigned short* qbf   = (unsigned short*)(ws + OFF_QBF);
  unsigned short* vt    = (unsigned short*)(ws + OFF_VT);
  unsigned short* wmat  = (unsigned short*)(ws + OFF_WMAT);
  float* geo4  = (float*)(ws + OFF_GEO4);
  float* gcoef = (float*)(ws + OFF_GCOEF);
  float* rsp   = (float*)(ws + OFF_RSP);

  k_cvt_fm <<<NBROW, 256, 0, stream>>>(fm, fmb);                    // 2048*1024 elems
  k_pack_w <<<dim3(32, 4, 3), 512, 0, stream>>>(Wk, Wq, Wv, wpack);
  k_geo    <<<NBROW / 256, 256, 0, stream>>>(boxes, geo4);
  k_gcoef  <<<1, 32, 0, stream>>>(Wg, bg, Wg2, bg2, gcoef);
  k_proj   <<<dim3(NBROW / 16, 4, 3), 32, 0, stream>>>(fmb, wpack, bk, bq, bv, kbf, qbf, vt);
  k_combine<<<dim3(NN / 16, NSPLIT, BATCH), 32, 0, stream>>>(kbf, qbf, geo4, gcoef, wmat, rsp);
  k_out    <<<dim3(NN / 16, 1, BATCH), 32, 0, stream>>>(wmat, vt, rsp, (float*)d_out);
}
